// GCN_71382356460176
// MI455X (gfx1250) — matrix operand order repaired and checked by reading, not compiled, not run on hardware
//
#include <hip/hip_runtime.h>
#include <hip/hip_bf16.h>
#include <stdint.h>

// ---------- types for WMMA ----------
typedef __attribute__((ext_vector_type(16))) __bf16 v16bf;
typedef __attribute__((ext_vector_type(8)))  float  v8f;

struct U128x2 { uint4 q0, q1; };            // 32 bytes == v16bf

__device__ __forceinline__ unsigned short f2bf(float f) {
    unsigned u = __float_as_uint(f);
    u += 0x7FFFu + ((u >> 16) & 1u);        // round-to-nearest-even
    return (unsigned short)(u >> 16);
}
__device__ __forceinline__ float bf2f(unsigned short s) {
    return __uint_as_float(((unsigned)s) << 16);
}

// ---------- generic fp32 -> bf16 convert (row-major copy) ----------
__global__ void cvt_bf16_kernel(const float* __restrict__ in, unsigned short* __restrict__ out,
                                int total) {
    int t = blockIdx.x * blockDim.x + threadIdx.x;
    if (t < total) out[t] = f2bf(in[t]);
}

// ---------- degree / normalization ----------
__global__ void deg_kernel(const int* __restrict__ dst, int E, float* __restrict__ deg) {
    int e = blockIdx.x * blockDim.x + threadIdx.x;
    if (e < E) atomicAdd(&deg[dst[e]], 1.0f);
}

__global__ void dis_kernel(float* __restrict__ deg, int N) {
    int i = blockIdx.x * blockDim.x + threadIdx.x;
    if (i < N) deg[i] = rsqrtf(deg[i] + 1.0f);   // +1 self-loop; deg+1 >= 1 always
}

// ---------- W transpose + fp32->bf16 convert: Wt[n*128 + k] = bf16(W[k*Ncol + n]) ----------
__global__ void wcvt_kernel(const float* __restrict__ W, int Ncol, int total,
                            unsigned short* __restrict__ Wt) {
    int t = blockIdx.x * blockDim.x + threadIdx.x;
    if (t >= total) return;
    int k = t / Ncol;
    int n = t - k * Ncol;
    Wt[n * 128 + k] = f2bf(W[t]);
}

// ---------- WMMA GEMM: Out[N x ldn](bf16) = A[N x 128](bf16) * Wt(bf16, [ldn][128]) ----------
// block = (32,4): one wave per 16x16 output tile. K fixed at 128 (4 x K=32 WMMA steps).
// A and B fragments are pure 16B vector loads; no per-tile conversion work.
__global__ void gemm_wmma_kernel(const unsigned short* __restrict__ A,
                                 const unsigned short* __restrict__ Wt,
                                 unsigned short* __restrict__ Out,
                                 int ldn) {
    const int lane = threadIdx.x;          // 0..31
    const int row  = lane & 15;            // A-matrix M within tile
    const int half = lane >> 4;            // lane group selects K sub-block
    const int ncol = lane & 15;            // B/C column within tile
    const int m0   = blockIdx.x * 16;
    const int n0   = (blockIdx.y * 4 + threadIdx.y) * 16;

    const unsigned short* arow = A  + (size_t)(m0 + row)  * 128;
    const unsigned short* bcol = Wt + (size_t)(n0 + ncol) * 128;

    v8f c = {0.f, 0.f, 0.f, 0.f, 0.f, 0.f, 0.f, 0.f};

#pragma unroll
    for (int k0 = 0; k0 < 128; k0 += 32) {
        // A fragment: elements 0..7  <- K = k0 + half*8 + (0..7)      (contiguous bf16)
        //             elements 8..15 <- K = k0 + 16 + half*8 + (0..7) (contiguous bf16)
        U128x2 aw;
        aw.q0 = *(const uint4*)(arow + k0 + half * 8);
        aw.q1 = *(const uint4*)(arow + k0 + 16 + half * 8);
        v16bf av = __builtin_bit_cast(v16bf, aw);

        // B fragment: lane = column, same K order as A (k0 + half*8 + 0..7, k0 + 16 + half*8 + 0..7)
        U128x2 bw;
        bw.q0 = *(const uint4*)(bcol + k0 + half * 8);
        bw.q1 = *(const uint4*)(bcol + k0 + 16 + half * 8);
        v16bf bv = __builtin_bit_cast(v16bf, bw);

        c = __builtin_amdgcn_wmma_f32_16x16x32_bf16(
                /*neg_a=*/false, av, /*neg_b=*/false, bv,
                /*c_mod=*/(short)0, c, /*reuse_a=*/false, /*reuse_b=*/false);
    }

    // C layout: VGPR r -> M = r + 8*half, N = lane&15
#pragma unroll
    for (int r = 0; r < 8; ++r) {
        int m = m0 + r + 8 * half;
        Out[(size_t)m * ldn + n0 + ncol] = f2bf(c[r]);
    }
}

// ---------- edge-parallel scatter-add: out[dst] += bf2f(h[src]) * dis[src]*dis[dst] ----------
// F = 1<<shift features per row; h and out share the same row stride F.
__global__ void agg_kernel(const int* __restrict__ src, const int* __restrict__ dst,
                           long long total, int shift,
                           const unsigned short* __restrict__ h,
                           const float* __restrict__ dis,
                           float* __restrict__ out) {
    long long t = (long long)blockIdx.x * blockDim.x + threadIdx.x;
    if (t >= total) return;
    int e = (int)(t >> shift);
    int f = (int)(t & ((1 << shift) - 1));
    int s = src[e];
    int d = dst[e];
    float norm = dis[s] * dis[d];
    float v = bf2f(h[((size_t)s << shift) + f]) * norm;
    atomicAdd(&out[((size_t)d << shift) + f], v);
}

// ---------- layer-1 epilogue: self-loop + bias + ReLU, emit bf16 A for layer 2 ----------
__global__ void relu_bias_kernel(const float* __restrict__ agg,
                                 const unsigned short* __restrict__ h,
                                 const float* __restrict__ dis,
                                 const float* __restrict__ bias,
                                 unsigned short* __restrict__ out_bf, int N) {
    int t = blockIdx.x * blockDim.x + threadIdx.x;   // N*128 threads
    int i = t >> 7;
    int f = t & 127;
    if (i >= N) return;
    float di = dis[i];
    float v = agg[t] + bf2f(h[t]) * di * di + bias[f];
    out_bf[t] = f2bf(fmaxf(v, 0.0f));
}

// ---------- final epilogue: self-loop + bias into d_out (F = 64) ----------
__global__ void final_epilogue_kernel(float* __restrict__ out,
                                      const unsigned short* __restrict__ h2,
                                      const float* __restrict__ dis,
                                      const float* __restrict__ bias, int N) {
    int t = blockIdx.x * blockDim.x + threadIdx.x;   // N*64 threads
    int i = t >> 6;
    int f = t & 63;
    if (i >= N) return;
    float di = dis[i];
    out[t] += bf2f(h2[t]) * di * di + bias[f];
}

static inline size_t align256(size_t x) { return (x + 255) & ~(size_t)255; }

extern "C" void kernel_launch(void* const* d_in, const int* in_sizes, int n_in,
                              void* d_out, int out_size, void* d_ws, size_t ws_size,
                              hipStream_t stream) {
    const float* x  = (const float*)d_in[0];
    const int*   ei = (const int*)d_in[1];
    // d_in[2] = batch (unused)
    const float* W1 = (const float*)d_in[3];
    const float* b1 = (const float*)d_in[4];
    const float* W2 = (const float*)d_in[5];
    const float* b2 = (const float*)d_in[6];

    const int N = in_sizes[0] / 128;   // 100000
    const int E = in_sizes[1] / 2;     // 1600000
    const int* src = ei;
    const int* dst = ei + E;
    float* out = (float*)d_out;

    // workspace carve-out
    char* w = (char*)d_ws;
    size_t o = 0;
    float*          deg  = (float*)(w + o);          o += align256((size_t)N * 4);
    unsigned short* xb   = (unsigned short*)(w + o); o += align256((size_t)N * 128 * 2); // bf16(x)
    unsigned short* h1   = (unsigned short*)(w + o); o += align256((size_t)N * 128 * 2); // bf16(x@W1)
    float*          agg1 = (float*)(w + o);          o += align256((size_t)N * 128 * 4); // fp32 accum
    unsigned short* a1b  = (unsigned short*)(w + o); o += align256((size_t)N * 128 * 2); // bf16 relu out
    unsigned short* h2   = (unsigned short*)(w + o); o += align256((size_t)N * 64 * 2);  // bf16(a1@W2)
    unsigned short* W1t  = (unsigned short*)(w + o); o += align256((size_t)128 * 128 * 2);
    unsigned short* W2t  = (unsigned short*)(w + o); o += align256((size_t)128 * 64 * 2);
    (void)ws_size; (void)n_in; (void)out_size;

    // zero accumulators (graph-capture-safe)
    hipMemsetAsync(deg,  0, (size_t)N * 4,       stream);
    hipMemsetAsync(agg1, 0, (size_t)N * 128 * 4, stream);
    hipMemsetAsync(out,  0, (size_t)N * 64 * 4,  stream);

    // one-time conversions: x -> bf16; weights -> bf16 transposed [Ncol][K=128]
    cvt_bf16_kernel<<<(N * 128 + 255) / 256, 256, 0, stream>>>(x, xb, N * 128);
    wcvt_kernel<<<(128 * 128 + 255) / 256, 256, 0, stream>>>(W1, 128, 128 * 128, W1t);
    wcvt_kernel<<<(128 * 64  + 255) / 256, 256, 0, stream>>>(W2, 64,  128 * 64,  W2t);

    // degrees + symmetric norm
    deg_kernel<<<(E + 255) / 256, 256, 0, stream>>>(dst, E, deg);
    dis_kernel<<<(N + 255) / 256, 256, 0, stream>>>(deg, N);

    // layer 1: h1 = bf16(x @ W1)
    {
        dim3 grid(N / 16, 2);     // 6250 x 2 ; threadIdx.y covers 4 col-tiles each
        dim3 block(32, 4);
        gemm_wmma_kernel<<<grid, block, 0, stream>>>(xb, W1t, h1, 128);
    }
    // scatter-add over edges (F=128)
    {
        long long total = (long long)E * 128;
        agg_kernel<<<(unsigned)((total + 255) / 256), 256, 0, stream>>>(
            src, dst, total, 7, h1, deg, agg1);
    }
    // self-loop + bias + ReLU -> bf16 operand for layer 2
    relu_bias_kernel<<<(N * 128 + 255) / 256, 256, 0, stream>>>(agg1, h1, deg, b1, a1b, N);

    // layer 2: h2 = bf16(a1b @ W2)
    {
        dim3 grid(N / 16, 1);     // threadIdx.y covers the 4 col-tiles of 64 outputs
        dim3 block(32, 4);
        gemm_wmma_kernel<<<grid, block, 0, stream>>>(a1b, W2t, h2, 64);
    }
    // scatter-add over edges (F=64) directly into d_out
    {
        long long total = (long long)E * 64;
        agg_kernel<<<(unsigned)((total + 255) / 256), 256, 0, stream>>>(
            src, dst, total, 6, h2, deg, out);
    }
    // self-loop + bias
    final_epilogue_kernel<<<(N * 64 + 255) / 256, 256, 0, stream>>>(out, h2, deg, b2, N);
}
